// MultiheadAttention_30691836297633
// MI455X (gfx1250) — matrix operand order repaired and checked by reading, not compiled, not run on hardware
//
#include <hip/hip_runtime.h>
#include <hip/hip_bf16.h>

// ---------------------------------------------------------------------------
// Multihead attention (B=2, L=2048, D=2048, H=16, Dh=128) for gfx1250.
// GEMMs: block-cooperative macro-tiles, TDM (tensor_load_to_lds) double
// buffering into LDS, v_wmma_f32_16x16x32_bf16 compute.
// ---------------------------------------------------------------------------

#define B_ 2
#define L_ 2048
#define D_ 2048
#define H_ 16
#define DH_ 128
#define QKV_N (3 * D_)   // 6144

typedef __attribute__((ext_vector_type(16))) __bf16 v16bf;
typedef __attribute__((ext_vector_type(8)))  float  v8f;
typedef __attribute__((ext_vector_type(4)))  unsigned int v4u;
typedef __attribute__((ext_vector_type(8)))  unsigned int v8u;

union ABf16 {
  v16bf v;
  unsigned int u[8];
  unsigned short s[16];
};

__device__ __forceinline__ unsigned short f32_to_bf16(float f) {
  unsigned int u = __float_as_uint(f);
  unsigned int r = u + 0x7FFFu + ((u >> 16) & 1u);   // round-to-nearest-even
  return (unsigned short)(r >> 16);
}

// A-matrix (16x32 bf16) element-pair K-base for VGPR v (0..7), half hi (0..1).
// Contiguous order (matches B): V0..7 hold K={0,1}..{14,15} (+16 for lanes 16-31).
__device__ __forceinline__ int a_pair_k(int v, int hi) {
  int base = 2 * v;
  return base + 16 * hi;
}

__device__ __forceinline__ float half_max16(float v) {
#pragma unroll
  for (int m = 1; m < 16; m <<= 1) v = fmaxf(v, __shfl_xor(v, m, 32));
  return v;
}
__device__ __forceinline__ float half_sum16(float v) {
#pragma unroll
  for (int m = 1; m < 16; m <<= 1) v += __shfl_xor(v, m, 32);
  return v;
}

// Low 32 bits of a flat pointer into LDS == LDS byte offset (ISA 10.2 aperture
// mapping).
__device__ __forceinline__ unsigned lds_off(const void* p) {
  return (unsigned)(unsigned long long)p;
}

// ---------------------------------------------------------------------------
// Tensor Data Mover: 2D tile load Global -> LDS (D# groups 0+1 only; VADDR2/3
// NULL => up-to-2D tensor form). Element size fixed at 2 bytes (bf16).
// Fields per CDNA5 ISA. Uniform args (SGPRs).
// ---------------------------------------------------------------------------
__device__ __forceinline__ void tdm_load_2d_bf16(
    unsigned lds_addr, const void* gaddr,
    unsigned tile_d0, unsigned tile_d1,
    unsigned tensor_d0, unsigned tensor_d1,
    unsigned long long stride_d0 /* elements */) {
  unsigned long long ga = (unsigned long long)gaddr;
  v4u g0;
  g0[0] = 1u;                                    // count=1, user descriptor
  g0[1] = lds_addr;                              // LDS byte address
  g0[2] = (unsigned)(ga & 0xFFFFFFFFu);          // global_addr[31:0]
  g0[3] = (unsigned)((ga >> 32) & 0x01FFFFFFu)   // global_addr[56:32]
        | (2u << 30);                            // type = 2 ("image")
  v8u g1;
  g1[0] = (1u << 16);                            // data_size=1 (2 bytes)
  g1[1] = (tensor_d0 << 16);                     // tensor_dim0[15:0] @bits63:48
  g1[2] = (tensor_d0 >> 16) | (tensor_d1 << 16); // dim0[31:16] | dim1[15:0]
  g1[3] = (tensor_d1 >> 16) | (tile_d0 << 16);   // dim1[31:16] | tile_dim0
  g1[4] = (tile_d1 & 0xFFFFu);                   // tile_dim1 | tile_dim2=0
  g1[5] = (unsigned)(stride_d0 & 0xFFFFFFFFu);   // dim0_stride[31:0]
  g1[6] = (unsigned)((stride_d0 >> 32) & 0xFFFFu); // stride[47:32]|dim1_str=0
  g1[7] = 0u;
  asm volatile("tensor_load_to_lds %0, %1" :: "s"(g0), "s"(g1) : "memory");
}

// ---------------------------------------------------------------------------
// Elementwise f32 -> bf16
// ---------------------------------------------------------------------------
__global__ __launch_bounds__(256) void conv_bf16_kernel(
    const float* __restrict__ in, unsigned short* __restrict__ out, int n) {
  int i = blockIdx.x * blockDim.x + threadIdx.x;
  if (i < n) out[i] = f32_to_bf16(in[i]);
}

// W: [K,N] f32 -> WT: [N,K] bf16 (so GEMM B-tiles read contiguous K per lane)
__global__ __launch_bounds__(256) void transpose_bf16_kernel(
    const float* __restrict__ W, unsigned short* __restrict__ WT, int K, int N) {
  int i = blockIdx.x * blockDim.x + threadIdx.x;
  if (i >= K * N) return;
  int n = i % N;
  int k = i / N;                         // coalesced read of W
  WT[(size_t)n * K + k] = f32_to_bf16(W[i]);
}

// ---------------------------------------------------------------------------
// GEMM: C[M,N] (f32) = A[M,K](bf16) * BT[N,K](bf16)^T (+ optional bias[N]).
// Block (8 waves) computes a 128(M) x 64(N) macro-tile. Per 32-wide K chunk:
// TDM stages A(128x32) and shared B(64x32) into LDS, double-buffered on
// TENSORcnt; each wave runs 4 WMMAs against the shared B tile.
// ---------------------------------------------------------------------------
__global__ __launch_bounds__(256) void gemm_bf16_wmma_kernel(
    const unsigned short* __restrict__ A,
    const unsigned short* __restrict__ BT,
    const float* __restrict__ bias,
    float* __restrict__ C,
    int M, int N, int K) {
  __shared__ __align__(16) unsigned short Abuf[2][128 * 32];  // 16KB
  __shared__ __align__(16) unsigned short Bbuf[2][64 * 32];   // 8KB

  const int lane = threadIdx.x & 31;
  const int w = threadIdx.x >> 5;        // wave 0..7 -> 16-row slice
  const int ntiles = N >> 6;
  const int mt = blockIdx.x / ntiles;    // 128-row macro tile
  const int nt = blockIdx.x % ntiles;    // 64-col macro tile
  const int hi = lane >> 4;
  const int ln = lane & 15;

  const unsigned short* Abase = A + (size_t)mt * 128 * K;
  const unsigned short* Bbase = BT + (size_t)nt * 64 * K;
  const int nk = K >> 5;

  v8f acc[4] = {};

  if (w == 0) {  // prologue: DMA chunk 0 into buffer 0
    tdm_load_2d_bf16(lds_off(&Abuf[0][0]), Abase, 32, 128,
                     (unsigned)K, (unsigned)M, (unsigned long long)K);
    tdm_load_2d_bf16(lds_off(&Bbuf[0][0]), Bbase, 32, 64,
                     (unsigned)K, (unsigned)N, (unsigned long long)K);
  }

  for (int i = 0; i < nk; ++i) {
    const int cur = i & 1;
    const int nxt = cur ^ 1;
    if (w == 0) {
      if (i + 1 < nk) {
        // prefetch next chunk, then wait for current (TENSORcnt is in-order)
        tdm_load_2d_bf16(lds_off(&Abuf[nxt][0]), Abase + (size_t)(i + 1) * 32,
                         32, 128, (unsigned)K, (unsigned)M,
                         (unsigned long long)K);
        tdm_load_2d_bf16(lds_off(&Bbuf[nxt][0]), Bbase + (size_t)(i + 1) * 32,
                         32, 64, (unsigned)K, (unsigned)N,
                         (unsigned long long)K);
        __builtin_amdgcn_s_wait_tensorcnt(2);
      } else {
        __builtin_amdgcn_s_wait_tensorcnt(0);
      }
    }
    __syncthreads();   // LDS buffer `cur` visible to all waves

    ABf16 a;
    const unsigned short* arow = &Abuf[cur][(w * 16 + ln) * 32];
#pragma unroll
    for (int v = 0; v < 8; ++v)
      a.u[v] = *(const unsigned int*)(arow + a_pair_k(v, hi));

#pragma unroll
    for (int j = 0; j < 4; ++j) {
      const unsigned short* bp = &Bbuf[cur][(j * 16 + ln) * 32 + 16 * hi];
      const uint4* p = (const uint4*)bp;
      uint4 lo = p[0];
      uint4 hh = p[1];
      ABf16 b;
      b.u[0] = lo.x; b.u[1] = lo.y; b.u[2] = lo.z; b.u[3] = lo.w;
      b.u[4] = hh.x; b.u[5] = hh.y; b.u[6] = hh.z; b.u[7] = hh.w;
      acc[j] = __builtin_amdgcn_wmma_f32_16x16x32_bf16(
          false, a.v, false, b.v, (short)0, acc[j], false, false);
    }
    __syncthreads();   // done reading `cur` before it is re-filled at i+2
  }

#pragma unroll
  for (int j = 0; j < 4; ++j) {
    int col = nt * 64 + j * 16 + ln;
    float bv = bias ? bias[col] : 0.0f;
#pragma unroll
    for (int r = 0; r < 8; ++r) {
      int row = mt * 128 + w * 16 + r + 8 * hi;  // C layout: VGPR r -> M row
      C[(size_t)row * N + col] = acc[j][r] + bv;
    }
  }
}

// ---------------------------------------------------------------------------
// RoPE + head split. qkv: [B*L, 6144] f32.
// Writes Q,K as [B*H, L, 128] bf16 and V transposed as [B*H, 128, L] bf16.
// ---------------------------------------------------------------------------
__global__ __launch_bounds__(256) void rope_split_kernel(
    const float* __restrict__ qkv,
    const float* __restrict__ fc,
    const float* __restrict__ fs,
    unsigned short* __restrict__ Qo,
    unsigned short* __restrict__ Ko,
    unsigned short* __restrict__ VTo) {
  int idx = blockIdx.x * blockDim.x + threadIdx.x;
  if (idx >= B_ * L_ * H_ * 64) return;
  int j = idx & 63;
  int h = (idx >> 6) & (H_ - 1);
  int l = (idx >> 10) & (L_ - 1);
  int b = idx >> 21;

  const float c = fc[l * 64 + j];
  const float s = fs[l * 64 + j];
  const size_t row = (size_t)(b * L_ + l) * QKV_N;
  const int col = h * DH_ + 2 * j;

  float qe = qkv[row + col],          qd = qkv[row + col + 1];
  float ke = qkv[row + D_ + col],     kd = qkv[row + D_ + col + 1];
  float ve = qkv[row + 2 * D_ + col], vd = qkv[row + 2 * D_ + col + 1];

  const size_t obase = ((size_t)(b * H_ + h) * L_ + l) * DH_ + 2 * j;
  Qo[obase]     = f32_to_bf16(qe * c - qd * s);
  Qo[obase + 1] = f32_to_bf16(qe * s + qd * c);
  Ko[obase]     = f32_to_bf16(ke * c - kd * s);
  Ko[obase + 1] = f32_to_bf16(ke * s + kd * c);

  const size_t vtb = ((size_t)(b * H_ + h) * DH_ + 2 * j) * L_ + l;
  VTo[vtb]      = f32_to_bf16(ve);
  VTo[vtb + L_] = f32_to_bf16(vd);
}

// ---------------------------------------------------------------------------
// Flash attention, causal. One wave (32 threads) per (b, h, 16-query tile).
// Key blocks of 32: S(16x32) via 8 WMMAs, online softmax with half-wave
// shuffle reductions, P transposed via LDS, O(16x128) via 8 WMMAs per block.
// ---------------------------------------------------------------------------
__global__ __launch_bounds__(32) void flash_attn_kernel(
    const unsigned short* __restrict__ Q,   // [B*H, L, 128]
    const unsigned short* __restrict__ Km,  // [B*H, L, 128]
    const unsigned short* __restrict__ VT,  // [B*H, 128, L]
    unsigned short* __restrict__ Obf) {     // [B, L, 2048]
  __shared__ __align__(16) unsigned short Plds[16 * 32];

  const int lane = threadIdx.x;
  const int hi = lane >> 4;
  const int ln = lane & 15;
  const int qtiles = L_ / 16;
  const int bh = blockIdx.x / qtiles;
  const int q0 = (blockIdx.x % qtiles) * 16;
  const int b = bh / H_;
  const int h = bh % H_;

  const unsigned short* qbase = Q  + (size_t)bh * L_ * DH_;
  const unsigned short* kbase = Km + (size_t)bh * L_ * DH_;
  const unsigned short* vbase = VT + (size_t)bh * DH_ * L_;

  ABf16 qa[4];
  {
    const unsigned short* qrow = qbase + (size_t)(q0 + ln) * DH_;
#pragma unroll
    for (int c = 0; c < 4; ++c)
#pragma unroll
      for (int v = 0; v < 8; ++v)
        qa[c].u[v] = *(const unsigned int*)(qrow + c * 32 + a_pair_k(v, hi));
  }

  float mrow[8], lrow[8];
  v8f O[8] = {};
#pragma unroll
  for (int r = 0; r < 8; ++r) { mrow[r] = -3.0e38f; lrow[r] = 0.0f; }

  const float scale = 0.08838834764831845f;  // 1/sqrt(128)
  const int nkb = (q0 + 16 + 31) >> 5;

  for (int kb = 0; kb < nkb; ++kb) {
    const int kstart = kb * 32;

    v8f S[2] = {};
#pragma unroll
    for (int x = 0; x < 2; ++x) {
      const unsigned short* krow = kbase + (size_t)(kstart + x * 16 + ln) * DH_;
#pragma unroll
      for (int c = 0; c < 4; ++c) {
        const uint4* p = (const uint4*)(krow + c * 32 + 16 * hi);
        uint4 lo = p[0];
        uint4 hh = p[1];
        ABf16 bm;
        bm.u[0] = lo.x; bm.u[1] = lo.y; bm.u[2] = lo.z; bm.u[3] = lo.w;
        bm.u[4] = hh.x; bm.u[5] = hh.y; bm.u[6] = hh.z; bm.u[7] = hh.w;
        S[x] = __builtin_amdgcn_wmma_f32_16x16x32_bf16(
            false, qa[c].v, false, bm.v, (short)0, S[x], false, false);
      }
    }

#pragma unroll
    for (int r = 0; r < 8; ++r) {
      const int query = q0 + r + 8 * hi;
#pragma unroll
      for (int x = 0; x < 2; ++x) {
        int key = kstart + x * 16 + ln;
        float sv = S[x][r] * scale;
        if (key > query) sv = -3.0e38f;
        S[x][r] = sv;
      }
      float mx = half_max16(fmaxf(S[0][r], S[1][r]));
      float mnew = fmaxf(mrow[r], mx);
      float alpha = __expf(mrow[r] - mnew);
      float p0 = __expf(S[0][r] - mnew);
      float p1 = __expf(S[1][r] - mnew);
      S[0][r] = p0;
      S[1][r] = p1;
      lrow[r] = lrow[r] * alpha + half_sum16(p0 + p1);
      mrow[r] = mnew;
#pragma unroll
      for (int ch = 0; ch < 8; ++ch) O[ch][r] = O[ch][r] * alpha;
    }

    __syncthreads();
#pragma unroll
    for (int r = 0; r < 8; ++r) {
      int row = r + 8 * hi;
      Plds[row * 32 + ln]      = f32_to_bf16(S[0][r]);
      Plds[row * 32 + 16 + ln] = f32_to_bf16(S[1][r]);
    }
    __syncthreads();
    ABf16 pa;
    {
      const unsigned short* prow = &Plds[ln * 32];
#pragma unroll
      for (int v = 0; v < 8; ++v)
        pa.u[v] = *(const unsigned int*)(prow + a_pair_k(v, hi));
    }
    __syncthreads();

#pragma unroll
    for (int ch = 0; ch < 8; ++ch) {
      const unsigned short* vrow =
          vbase + (size_t)(ch * 16 + ln) * L_ + kstart + 16 * hi;
      const uint4* p = (const uint4*)vrow;
      uint4 lo = p[0];
      uint4 hh = p[1];
      ABf16 bm;
      bm.u[0] = lo.x; bm.u[1] = lo.y; bm.u[2] = lo.z; bm.u[3] = lo.w;
      bm.u[4] = hh.x; bm.u[5] = hh.y; bm.u[6] = hh.z; bm.u[7] = hh.w;
      O[ch] = __builtin_amdgcn_wmma_f32_16x16x32_bf16(
          false, pa.v, false, bm.v, (short)0, O[ch], false, false);
    }
  }

#pragma unroll
  for (int ch = 0; ch < 8; ++ch) {
    int col = h * DH_ + ch * 16 + ln;
#pragma unroll
    for (int r = 0; r < 8; ++r) {
      int row = r + 8 * hi;
      float val = O[ch][r] / lrow[r];
      Obf[(size_t)(b * L_ + q0 + row) * D_ + col] = f32_to_bf16(val);
    }
  }
}

// ---------------------------------------------------------------------------
// Host-side orchestration
// ---------------------------------------------------------------------------
extern "C" void kernel_launch(void* const* d_in, const int* in_sizes, int n_in,
                              void* d_out, int out_size, void* d_ws, size_t ws_size,
                              hipStream_t stream) {
  (void)in_sizes; (void)n_in; (void)out_size; (void)ws_size;

  const float* x    = (const float*)d_in[0];   // [B,L,D]
  const float* fcos = (const float*)d_in[1];   // [L,64]
  const float* fsin = (const float*)d_in[2];   // [L,64]
  const float* Wqkv = (const float*)d_in[3];   // [D, 3D]
  const float* Wc   = (const float*)d_in[4];   // [D, D]
  const float* bc   = (const float*)d_in[5];   // [D]
  float* out = (float*)d_out;                  // [B,L,D]

  char* ws = (char*)d_ws;
  size_t off = 0;
  auto carve = [&](size_t bytes) -> char* {
    char* p = ws + off;
    off = (off + bytes + 255) & ~(size_t)255;
    return p;
  };
  const size_t nx    = (size_t)B_ * L_ * D_;
  const size_t nqkvw = (size_t)D_ * QKV_N;
  const size_t nwc   = (size_t)D_ * D_;
  const size_t nqkv  = (size_t)B_ * L_ * QKV_N;
  const size_t nhead = (size_t)B_ * H_ * L_ * DH_;

  unsigned short* x_bf    = (unsigned short*)carve(nx * 2);
  unsigned short* WqkvT   = (unsigned short*)carve(nqkvw * 2);
  unsigned short* WcT     = (unsigned short*)carve(nwc * 2);
  float*          qkv_f32 = (float*)carve(nqkv * 4);
  unsigned short* Qh      = (unsigned short*)carve(nhead * 2);
  unsigned short* Kh      = (unsigned short*)carve(nhead * 2);
  unsigned short* VTh     = (unsigned short*)carve(nhead * 2);
  unsigned short* attn_bf = (unsigned short*)carve(nx * 2);

  // 1. precision conversion / weight transposes
  conv_bf16_kernel<<<(int)((nx + 255) / 256), 256, 0, stream>>>(x, x_bf, (int)nx);
  transpose_bf16_kernel<<<(int)((nqkvw + 255) / 256), 256, 0, stream>>>(
      Wqkv, WqkvT, D_, QKV_N);
  transpose_bf16_kernel<<<(int)((nwc + 255) / 256), 256, 0, stream>>>(
      Wc, WcT, D_, D_);

  // 2. QKV projection: [4096, 6144] = x_bf @ Wqkv (TDM-staged WMMA GEMM)
  {
    int M = B_ * L_, N = QKV_N, K = D_;
    int blocks = (M / 128) * (N / 64);
    gemm_bf16_wmma_kernel<<<blocks, 256, 0, stream>>>(
        x_bf, WqkvT, nullptr, qkv_f32, M, N, K);
  }

  // 3. RoPE + head split (+ V transpose)
  {
    int total = B_ * L_ * H_ * 64;
    rope_split_kernel<<<(total + 255) / 256, 256, 0, stream>>>(
        qkv_f32, fcos, fsin, Qh, Kh, VTh);
  }

  // 4. causal flash attention
  {
    int blocks = B_ * H_ * (L_ / 16);
    flash_attn_kernel<<<blocks, 32, 0, stream>>>(Qh, Kh, VTh, attn_bf);
  }

  // 5. output projection: out = attn @ Wc + bc
  {
    int M = B_ * L_, N = D_, K = D_;
    int blocks = (M / 128) * (N / 64);
    gemm_bf16_wmma_kernel<<<blocks, 256, 0, stream>>>(
        attn_bf, WcT, bc, out, M, N, K);
  }
}
